// S4Block_90941637525627
// MI455X (gfx1250) — matrix operand order repaired and checked by reading, not compiled, not run on hardware
//
#include <hip/hip_runtime.h>
#include <hip/hip_bf16.h>
#include <math.h>

// Problem sizes (fixed by reference)
#define BB 8
#define LL 4096
#define HH 512
#define NN 32
#define MM (BB * LL)   // 32768 rows

typedef __bf16 bf16;
typedef __bf16 v16bf __attribute__((ext_vector_type(16)));
typedef float  v8f   __attribute__((ext_vector_type(8)));

__device__ __forceinline__ float gelu_exact(float x) {
    return 0.5f * x * (1.f + erff(x * 0.70710678118654752f));
}

// ---------------------------------------------------------------------------
// 1) Per-(h,n) SSM constants: lambda = exp(dt*A), C' = C*(exp(dtA)-1)/A, x2
//    folded in for the conjugate-pair sum.
// ---------------------------------------------------------------------------
__global__ __launch_bounds__(256)
void prep_consts_kernel(const float* __restrict__ log_dt,
                        const float* __restrict__ logA_re,
                        const float* __restrict__ A_im,
                        const float* __restrict__ C_re,
                        const float* __restrict__ C_im,
                        float* __restrict__ lamr, float* __restrict__ lami,
                        float* __restrict__ c2r,  float* __restrict__ c2i) {
    int idx = blockIdx.x * 256 + threadIdx.x;
    if (idx >= HH * NN) return;
    int h = idx >> 5;
    float dt  = expf(log_dt[h]);
    float Are = -expf(logA_re[idx]);
    float Aim = A_im[idx];
    float dr = dt * Are, di = dt * Aim;
    float er = expf(dr);
    float lr = er * cosf(di);
    float li = er * sinf(di);
    float e1r = lr - 1.f, e1i = li;                    // exp(dtA) - 1
    float inv = 1.f / (Are * Are + Aim * Aim);
    float wr = (e1r * Are + e1i * Aim) * inv;          // (e-1)/A
    float wi = (e1i * Are - e1r * Aim) * inv;
    float cr = C_re[idx], ci = C_im[idx];
    lamr[idx] = lr;
    lami[idx] = li;
    c2r[idx] = 2.f * (cr * wr - ci * wi);
    c2i[idx] = 2.f * (cr * wi + ci * wr);
}

// ---------------------------------------------------------------------------
// 2) Weight convert fp32 [N][K] -> bf16 [N][K] (no transpose needed: for
//    y @ W.T the GEMM B operand B[k][n] = W[n][k], and the B-fragment layout
//    wants [n][k]-major anyway). ldb/kofs let out_W / proj_W concatenate
//    along K for the fused final GEMM.
// ---------------------------------------------------------------------------
__global__ __launch_bounds__(256)
void wcvt_kernel(const float* __restrict__ W, bf16* __restrict__ Wb,
                 int ldb, int kofs) {
    int e = (blockIdx.x * 256 + threadIdx.x) * 8;      // 512*512 elements
    int n = e >> 9;
    int k = e & (HH - 1);
    const float* s = W + (size_t)n * HH + k;
    bf16 tmp[8];
#pragma unroll
    for (int j = 0; j < 8; ++j) tmp[j] = (bf16)s[j];
    *(uint4*)&Wb[(size_t)n * ldb + kofs + k] = *(const uint4*)tmp;
}

// ---------------------------------------------------------------------------
// 3) LayerNorm over H=512: one wave per row, 16 elems/lane, shfl_xor reduce.
// ---------------------------------------------------------------------------
__global__ __launch_bounds__(256)
void ln_kernel(const float* __restrict__ x, const float* __restrict__ g,
               const float* __restrict__ beta,
               float* __restrict__ outf, bf16* __restrict__ outb) {
    int wave = threadIdx.x >> 5, lane = threadIdx.x & 31;
    size_t row = (size_t)blockIdx.x * 8 + wave;
    const float* xr = x + row * HH;
    float v[16];
    float s = 0.f;
#pragma unroll
    for (int i = 0; i < 16; ++i) { v[i] = xr[i * 32 + lane]; s += v[i]; }
#pragma unroll
    for (int off = 16; off >= 1; off >>= 1) s += __shfl_xor(s, off, 32);
    float mu = s * (1.f / 512.f);
    float q = 0.f;
#pragma unroll
    for (int i = 0; i < 16; ++i) { float d = v[i] - mu; q = fmaf(d, d, q); }
#pragma unroll
    for (int off = 16; off >= 1; off >>= 1) q += __shfl_xor(q, off, 32);
    float rstd = rsqrtf(q * (1.f / 512.f) + 1e-5f);
#pragma unroll
    for (int i = 0; i < 16; ++i) {
        int h = i * 32 + lane;
        float o = (v[i] - mu) * rstd * g[h] + beta[h];
        if (outf) outf[row * HH + h] = o;
        if (outb) outb[row * HH + h] = (bf16)o;
    }
}

// ---------------------------------------------------------------------------
// 4) Transpose fp32 [B][L][H] -> [B][H][L]
// ---------------------------------------------------------------------------
__global__ __launch_bounds__(256)
void transpose_f32_kernel(const float* __restrict__ src, float* __restrict__ dst) {
    __shared__ float tile[32][33];
    int b = blockIdx.z;
    int h0 = blockIdx.x * 32, l0 = blockIdx.y * 32;
    int tx = threadIdx.x, ty = threadIdx.y;
    for (int i = ty; i < 32; i += 8)
        tile[i][tx] = src[((size_t)b * LL + l0 + i) * HH + h0 + tx];
    __syncthreads();
    for (int i = ty; i < 32; i += 8)
        dst[((size_t)b * HH + h0 + i) * LL + l0 + tx] = tile[tx][i];
}

// ---------------------------------------------------------------------------
// 5) SSM scan: one wave32 per (b,h); lane n holds the complex state of mode n.
//    Every 32 steps: lanes dump per-step partials to LDS ([n][t], pad 33 ->
//    conflict-free both ways), lane t reduces over n, adds D*u, exact GELU,
//    coalesced store into yT[b][h][l].
// ---------------------------------------------------------------------------
__global__ __launch_bounds__(256)
void scan_kernel(const float* __restrict__ uT,
                 const float* __restrict__ lamr_g, const float* __restrict__ lami_g,
                 const float* __restrict__ c2r_g,  const float* __restrict__ c2i_g,
                 const float* __restrict__ Dparam,
                 float* __restrict__ yT) {
    __shared__ float ubuf[8][32];
    __shared__ float pbuf[8][32][33];
    int wave = threadIdx.x >> 5, lane = threadIdx.x & 31;
    int bh = blockIdx.x * 8 + wave;            // b*512 + h
    int h = bh & (HH - 1);
    float lr = lamr_g[h * NN + lane], li = lami_g[h * NN + lane];
    float cr = c2r_g[h * NN + lane], ci = c2i_g[h * NN + lane];
    float Dh = Dparam[h];
    const float* u = uT + (size_t)bh * LL;
    float* y = yT + (size_t)bh * LL;
    float sre = 0.f, sim = 0.f;
    for (int l0 = 0; l0 < LL; l0 += 32) {
        ubuf[wave][lane] = u[l0 + lane];
#pragma unroll 4
        for (int t = 0; t < 32; ++t) {
            float ut = ubuf[wave][t];                              // LDS broadcast
            float nre = fmaf(lr, sre, fmaf(-li, sim, ut));
            float nim = fmaf(lr, sim, li * sre);
            sre = nre; sim = nim;
            pbuf[wave][lane][t] = fmaf(cr, sre, -ci * sim);        // 2*Re(C'*s)
        }
        float acc = 0.f;
#pragma unroll
        for (int n = 0; n < 32; ++n) acc += pbuf[wave][n][lane];   // conflict-free
        float ut = ubuf[wave][lane];
        float val = fmaf(Dh, ut, acc);
        y[l0 + lane] = gelu_exact(val);
    }
}

// ---------------------------------------------------------------------------
// 6) Transpose fp32 [B][H][L] -> bf16 [B][L][H]
// ---------------------------------------------------------------------------
__global__ __launch_bounds__(256)
void transpose_to_bf_kernel(const float* __restrict__ src, bf16* __restrict__ dst) {
    __shared__ float tile[32][33];
    int b = blockIdx.z;
    int l0 = blockIdx.x * 32, h0 = blockIdx.y * 32;
    int tx = threadIdx.x, ty = threadIdx.y;
    for (int i = ty; i < 32; i += 8)
        tile[i][tx] = src[((size_t)b * HH + h0 + i) * LL + l0 + tx];
    __syncthreads();
    for (int i = ty; i < 32; i += 8)
        dst[((size_t)b * LL + l0 + i) * HH + h0 + tx] = (bf16)tile[tx][i];
}

// ---------------------------------------------------------------------------
// 7) bf16 WMMA GEMM: C[M,512] = A[M,K] @ W[N,K]^T, fp32 accumulate.
//    Block tile 128x64, 8 waves (wave32), each wave 32x32 via 2x2
//    v_wmma_f32_16x16x32_bf16 frags. LDS: A as [m][k], B as [n][k]
//    (pad 40 -> 16B-aligned rows). Both tiles staged with b128 load +
//    b128 LDS store per thread; fragment fills read k = 16*hi + s
//    for A and B alike (contiguous 16B segments, same k order for both).
//    MODE 0: +bias, +fp32 residual -> fp32 out AND bf16 aux copy (strided)
//    MODE 1: +bias, exact GELU -> bf16 out (strided)
//    MODE 2: +bias0+bias1 -> fp32 out (final)
// ---------------------------------------------------------------------------
template <int MODE>
__global__ __launch_bounds__(256)
void gemm_kernel(const bf16* __restrict__ A, int lda,
                 const bf16* __restrict__ Bw, int ldb, int K,
                 const float* __restrict__ bias0,
                 const float* __restrict__ bias1,
                 const float* __restrict__ resid,
                 float* __restrict__ outf,
                 bf16* __restrict__ outb, int ldo_b) {
    __shared__ __align__(16) bf16 As[128][40];
    __shared__ __align__(16) bf16 Bs[64][40];
    int tid = threadIdx.x;
    int lane = tid & 31, wave = tid >> 5;
    int wm = wave & 3, wn = wave >> 2;          // wave grid 4x2 over 128x64
    int m0 = blockIdx.y * 128, n0 = blockIdx.x * 64;
    int mlo = lane & 15, hi = lane >> 4;

    v8f c[2][2];
#pragma unroll
    for (int i = 0; i < 2; ++i)
#pragma unroll
        for (int j = 0; j < 2; ++j)
#pragma unroll
            for (int v = 0; v < 8; ++v) c[i][j][v] = 0.f;

    for (int kk = 0; kk < K; kk += 32) {
        __syncthreads();
        {   // A tile: 128 rows x 32 k; thread: 16 bf16 (one b128 pair)
            int r = tid >> 1, hlf = tid & 1;
            const bf16* src = A + (size_t)(m0 + r) * lda + kk + hlf * 16;
            if (kk + 32 < K) __builtin_prefetch(src + 32, 0, 0);
            const uint4* s4 = (const uint4*)src;
            uint4 a0 = s4[0], a1 = s4[1];
            uint4* d4 = (uint4*)&As[r][hlf * 16];
            d4[0] = a0; d4[1] = a1;
        }
        {   // B tile: 64 n-rows x 32 k; thread: 8 bf16, b128 in + b128 to LDS
            int n = tid >> 2, seg = tid & 3;
            const bf16* src = Bw + (size_t)(n0 + n) * ldb + kk + seg * 8;
            if (kk + 32 < K) __builtin_prefetch(src + 32, 0, 0);
            uint4 b0 = *(const uint4*)src;
            *(uint4*)&Bs[n][seg * 8] = b0;
        }
        __syncthreads();

        v16bf af[2], bf_[2];
#pragma unroll
        for (int i = 0; i < 2; ++i) {
            int m = wm * 32 + i * 16 + mlo;
            const bf16* p = &As[m][hi * 16];       // K = hi*16 + 0..7
            const bf16* q = &As[m][hi * 16 + 8];   // K = hi*16 + 8..15
#pragma unroll
            for (int j = 0; j < 8; ++j) { af[i][j] = p[j]; af[i][8 + j] = q[j]; }
        }
#pragma unroll
        for (int i = 0; i < 2; ++i) {
            int n = wn * 32 + i * 16 + mlo;
            const bf16* p = &Bs[n][hi * 16];       // K = hi*16 + 0..15
#pragma unroll
            for (int j = 0; j < 16; ++j) bf_[i][j] = p[j];
        }
#pragma unroll
        for (int i = 0; i < 2; ++i)
#pragma unroll
            for (int j = 0; j < 2; ++j)
                c[i][j] = __builtin_amdgcn_wmma_f32_16x16x32_bf16(
                    false, af[i], false, bf_[j], (short)0, c[i][j], false, false);
    }

    // Epilogue: C layout = VGPR v -> row (hi*8+v), col (lane&15)
#pragma unroll
    for (int i = 0; i < 2; ++i) {
#pragma unroll
        for (int j = 0; j < 2; ++j) {
            int col = n0 + wn * 32 + j * 16 + mlo;
            float bs = bias0[col];
            if (MODE == 2) bs += bias1[col];
#pragma unroll
            for (int v = 0; v < 8; ++v) {
                int row = m0 + wm * 32 + i * 16 + hi * 8 + v;
                float val = c[i][j][v] + bs;
                if (MODE == 0) {
                    val += resid[(size_t)row * HH + col];
                    outf[(size_t)row * HH + col] = val;
                    outb[(size_t)row * ldo_b + col] = (bf16)val;
                } else if (MODE == 1) {
                    val = gelu_exact(val);
                    outb[(size_t)row * ldo_b + col] = (bf16)val;
                } else {
                    outf[(size_t)row * HH + col] = val;
                }
            }
        }
    }
}

// ---------------------------------------------------------------------------
// 8) Zero the returned state tensor [B,H,N]
// ---------------------------------------------------------------------------
__global__ __launch_bounds__(256)
void zero_kernel(float* __restrict__ p, int n) {
    int i = blockIdx.x * 256 + threadIdx.x;
    if (i < n) p[i] = 0.f;
}

// ---------------------------------------------------------------------------
extern "C" void kernel_launch(void* const* d_in, const int* in_sizes, int n_in,
                              void* d_out, int out_size, void* d_ws, size_t ws_size,
                              hipStream_t stream) {
    const float* input   = (const float*)d_in[0];
    const float* ln1_g   = (const float*)d_in[1];
    const float* ln1_b   = (const float*)d_in[2];
    const float* log_dt  = (const float*)d_in[3];
    const float* logA_re = (const float*)d_in[4];
    const float* A_im    = (const float*)d_in[5];
    const float* C_re    = (const float*)d_in[6];
    const float* C_im    = (const float*)d_in[7];
    const float* Dp      = (const float*)d_in[8];
    const float* lin1_W  = (const float*)d_in[9];
    const float* lin1_b  = (const float*)d_in[10];
    const float* ln2_g   = (const float*)d_in[11];
    const float* ln2_b   = (const float*)d_in[12];
    const float* lin_W   = (const float*)d_in[13];
    const float* lin_b   = (const float*)d_in[14];
    const float* out_W   = (const float*)d_in[15];
    const float* out_b   = (const float*)d_in[16];
    const float* proj_W  = (const float*)d_in[17];
    const float* proj_b  = (const float*)d_in[18];

    char* base = (char*)d_ws;
    size_t off = 0;
    auto alloc = [&](size_t bytes) -> void* {
        void* p = base + off;
        off += (bytes + 255) & ~(size_t)255;
        return p;
    };
    float* lamr = (float*)alloc(HH * NN * 4);
    float* lami = (float*)alloc(HH * NN * 4);
    float* c2r  = (float*)alloc(HH * NN * 4);
    float* c2i  = (float*)alloc(HH * NN * 4);
    bf16* wb1    = (bf16*)alloc((size_t)HH * HH * 2);      // lin1_W bf16 [N][K]
    bf16* wb_lin = (bf16*)alloc((size_t)HH * HH * 2);      // lin_W  bf16 [N][K]
    bf16* wb_cat = (bf16*)alloc((size_t)2 * HH * HH * 2);  // [N][1024]: out_W | proj_W
    float* bufA = (float*)alloc((size_t)MM * HH * 4);      // x1, later y_res (fp32)
    float* bufB = (float*)alloc((size_t)MM * HH * 4);      // x1T, later fused bf16 [M,1024]
    float* bufC = (float*)alloc((size_t)MM * HH * 4);      // yT, later z_ln (bf16)
    bf16*  y_bf = (bf16*)alloc((size_t)MM * HH * 2);       // conv output, bf16 [M,512]

    float* x1    = bufA;
    float* x1T   = bufB;
    float* yT    = bufC;
    float* y_res = bufA;                 // reuse: x1 dead after transpose
    bf16*  fused = (bf16*)bufB;          // reuse: x1T dead after scan; [M,1024]
    bf16*  z_ln  = (bf16*)bufC;          // reuse: yT dead after transpose-to-bf

    float* zout  = (float*)d_out;
    float* state = zout + (size_t)MM * HH;

    // 1. SSM constants
    prep_consts_kernel<<<(HH * NN + 255) / 256, 256, 0, stream>>>(
        log_dt, logA_re, A_im, C_re, C_im, lamr, lami, c2r, c2i);

    // 2. Weight conversion fp32 -> bf16 (layout preserved: [N][K])
    int wblocks = (HH * HH) / (256 * 8);   // 128
    wcvt_kernel<<<wblocks, 256, 0, stream>>>(lin1_W, wb1, HH, 0);
    wcvt_kernel<<<wblocks, 256, 0, stream>>>(lin_W, wb_lin, HH, 0);
    wcvt_kernel<<<wblocks, 256, 0, stream>>>(out_W, wb_cat, 2 * HH, 0);
    wcvt_kernel<<<wblocks, 256, 0, stream>>>(proj_W, wb_cat, 2 * HH, HH);

    // 3. LN1 -> x1 (fp32)
    ln_kernel<<<MM / 8, 256, 0, stream>>>(input, ln1_g, ln1_b, x1, (bf16*)nullptr);

    // 4. x1 [B,L,H] -> x1T [B,H,L]
    transpose_f32_kernel<<<dim3(HH / 32, LL / 32, BB), dim3(32, 8), 0, stream>>>(x1, x1T);

    // 5. SSM scan + D skip + GELU -> yT [B,H,L]
    scan_kernel<<<(BB * HH) / 8, 256, 0, stream>>>(x1T, lamr, lami, c2r, c2i, Dp, yT);

    // 6. yT -> y_bf [B,L,H] bf16
    transpose_to_bf_kernel<<<dim3(LL / 32, HH / 32, BB), dim3(32, 8), 0, stream>>>(yT, y_bf);

    // 7. GEMM1: y_res = y_bf @ lin1_W^T + lin1_b + input ; bf16 copy into fused[:,512:]
    dim3 gg(HH / 64, MM / 128);
    gemm_kernel<0><<<gg, 256, 0, stream>>>(y_bf, HH, wb1, HH, HH, lin1_b, nullptr,
                                           input, y_res, fused + HH, 2 * HH);

    // 8. LN2 -> z_ln bf16
    ln_kernel<<<MM / 8, 256, 0, stream>>>(y_res, ln2_g, ln2_b, (float*)nullptr, z_ln);

    // 9. GEMM2: fused[:,0:512] = gelu(z_ln @ lin_W^T + lin_b)
    gemm_kernel<1><<<gg, 256, 0, stream>>>(z_ln, HH, wb_lin, HH, HH, lin_b, nullptr,
                                           nullptr, nullptr, fused, 2 * HH);

    // 10. GEMM3 (fused dual-branch, K=1024):
    //     z = [t | y_res]_bf16 @ [out_W^T ; proj_W^T] + out_b + proj_b
    gemm_kernel<2><<<gg, 256, 0, stream>>>(fused, 2 * HH, wb_cat, 2 * HH, 2 * HH,
                                           out_b, proj_b, nullptr, zout, nullptr, 0);

    // 11. state output = zeros [B,H,N]
    zero_kernel<<<(BB * HH * NN + 255) / 256, 256, 0, stream>>>(state, BB * HH * NN);
}
